// GptOssTopKRouter_11424613007750
// MI455X (gfx1250) — matrix operand order repaired and checked by reading, not compiled, not run on hardware
//
#include <hip/hip_runtime.h>
#include <cstdint>
#include <cstddef>

// CDNA5 WMMA vector types
typedef __attribute__((ext_vector_type(16))) __bf16 v16bf;
typedef __attribute__((ext_vector_type(8)))  float  v8f;
typedef __attribute__((ext_vector_type(8)))  unsigned v8u;
typedef __attribute__((ext_vector_type(4)))  unsigned v4u;
typedef __attribute__((ext_vector_type(8)))  int      v8i;
typedef __attribute__((ext_vector_type(4)))  int      v4i;

#define BM 64        // tokens per workgroup
#define BN 128       // experts (full width)
#define KB 64        // K-chunk (elements)
#define KB2 (KB / 2) // packed bf16 pairs (uints) per chunk row

struct Frag32 { uint4 lo, hi; }; // 32 bytes -> bit_cast to v16bf

// fp32 -> packed 2x bf16: round-half-up + one v_perm_b32 (3 VALU per pair)
__device__ __forceinline__ unsigned pk_bf16(float x, float y) {
#if __has_builtin(__builtin_amdgcn_cvt_pk_bf16_f32)
  auto r = __builtin_amdgcn_cvt_pk_bf16_f32(x, y);
  return __builtin_bit_cast(unsigned, r);
#else
  unsigned a = __builtin_bit_cast(unsigned, x) + 0x8000u;
  unsigned b = __builtin_bit_cast(unsigned, y) + 0x8000u;
  return __builtin_amdgcn_perm(b, a, 0x07060302u);
#endif
}

__device__ __forceinline__ uint4 pk4(const float4 a, const float4 b) {
  uint4 r;
  r.x = pk_bf16(a.x, a.y);
  r.y = pk_bf16(a.z, a.w);
  r.z = pk_bf16(b.x, b.y);
  r.w = pk_bf16(b.z, b.w);
  return r;
}

// ---------------- shared fused top-4 + softmax + scatter epilogue ----------------
__device__ __forceinline__ void topk_epilogue(const float (*logits)[BN],
                                              float* __restrict__ out,
                                              int rowBase, int w, int lane) {
  for (int t = w * 8; t < w * 8 + 8; ++t) {
    const int e0 = lane * 4;                        // lane owns experts e0..e0+3
    float4 lv = *(const float4*)&logits[t][e0];
    float v[4] = {lv.x, lv.y, lv.z, lv.w};
    float tv[4]; int ti[4];
#pragma unroll
    for (int rsel = 0; rsel < 4; ++rsel) {
      float bvv = v[0]; int bi = e0;
#pragma unroll
      for (int j = 1; j < 4; ++j)
        if (v[j] > bvv || (v[j] == bvv && e0 + j < bi)) { bvv = v[j]; bi = e0 + j; }
#pragma unroll
      for (int off = 16; off > 0; off >>= 1) {       // wave32 butterfly argmax
        float ov = __shfl_xor(bvv, off, 32);
        int   oi = __shfl_xor(bi,  off, 32);
        if (ov > bvv || (ov == bvv && oi < bi)) { bvv = ov; bi = oi; }
      }
      tv[rsel] = bvv; ti[rsel] = bi;
#pragma unroll
      for (int j = 0; j < 4; ++j)
        if (e0 + j == bi) v[j] = -__builtin_inff();
    }
    float m = tv[0], s = 0.0f, p[4];
#pragma unroll
    for (int rsel = 0; rsel < 4; ++rsel) { p[rsel] = __expf(tv[rsel] - m); s += p[rsel]; }
    float inv = 1.0f / s;
    float o[4] = {0.0f, 0.0f, 0.0f, 0.0f};
#pragma unroll
    for (int j = 0; j < 4; ++j)
#pragma unroll
      for (int rsel = 0; rsel < 4; ++rsel)
        if (e0 + j == ti[rsel]) o[j] = p[rsel] * inv;
    *(float4*)&out[(size_t)(rowBase + t) * BN + e0] = make_float4(o[0], o[1], o[2], o[3]);
  }
}

// =======================================================================
// W pre-convert: fp32 -> bf16 once (0.7 MB, stays L2-resident afterwards)
// =======================================================================
__global__ __launch_bounds__(256) void cvt_w_kernel(const float* __restrict__ Wt,
                                                    unsigned* __restrict__ Wb, int n8) {
  int i = blockIdx.x * 256 + threadIdx.x;   // each thread: 8 floats -> 4 packed uints
  if (i < n8) {
    const float4* p = (const float4*)Wt + (size_t)i * 2;
    float4 x = p[0], y = p[1];
    *((uint4*)Wb + i) = pk4(x, y);
  }
}

// =======================================================================
// Main kernel, TDM variant: B tiles DMA'd bf16 global->LDS (double buffered)
// =======================================================================
union SMemT {
  struct {
    alignas(16) unsigned a[BM * KB2];        //  8 KB bf16 A tile
    alignas(16) unsigned b[2][BN * KB2];     //  2 x 16 KB bf16 B tiles (TDM dest)
  } g;
  alignas(16) float logits[BM][BN];          // 32 KB fp32 logits overlay
};

__global__ __launch_bounds__(256) void router_topk_tdm_kernel(
    const float* __restrict__ Hs,     // [T, Hd] fp32
    const unsigned* __restrict__ Wb,  // [BN, Hd] bf16 (packed pairs)
    const float* __restrict__ bias,   // [BN]
    float* __restrict__ out,          // [T, BN]
    int Hd)
{
  __shared__ SMemT sm;
  const int tid  = threadIdx.x;
  const int lane = tid & 31;
  const int w    = tid >> 5;
  const int wm   = w & 1;
  const int wn   = w >> 1;
  const int rowBase = blockIdx.x * BM;

  v8f acc[2][2] = {};
  float4 ra[2][2];                    // staged A fp32 (software pipeline)
  const int nCh = Hd / KB;

  // Issue one TENSOR_LOAD_TO_LDS: 2D D#, 128x64 bf16 tile, row stride Hd.
  auto tdm_issue = [&](int kk, int buf) {
    unsigned lds_off = (unsigned)(uintptr_t)(&sm.g.b[buf][0]); // LDS = flat addr[31:0]
    unsigned long long ga = (unsigned long long)(uintptr_t)Wb + (unsigned long long)kk * 2ull;
    v4u g0;
    g0[0] = 1u;                                           // count=1, user descriptor
    g0[1] = lds_off;                                      // lds_addr
    g0[2] = (unsigned)(ga & 0xffffffffull);               // global_addr[31:0]
    g0[3] = (unsigned)((ga >> 32) & 0x1ffffffull)         // global_addr[56:32]
            | (2u << 30);                                 // type=2 (image)
    v8u g1 = (v8u)0;
    g1[0] = 0x00010000u;                                  // data_size=1 (2 bytes/elem)
    g1[1] = ((unsigned)Hd & 0xffffu) << 16;               // tensor_dim0[15:0]
    g1[2] = ((unsigned)Hd >> 16) | ((unsigned)BN << 16);  // dim0[31:16] | tensor_dim1[15:0]
    g1[3] = ((unsigned)KB << 16);                         // dim1[31:16]=0 | tile_dim0=KB
    g1[4] = (unsigned)BN;                                 // tile_dim1=BN, tile_dim2=0
    g1[5] = (unsigned)Hd;                                 // tensor_dim0_stride[31:0]
    // g1[6], g1[7] = 0: stride0[47:32]=0, tensor_dim1_stride=0 (unused, 2D)
    v4i z4 = (v4i)0;
#if defined(__clang_major__) && (__clang_major__ >= 23)
    v8i z8 = (v8i)0;
    __builtin_amdgcn_tensor_load_to_lds(g0, __builtin_bit_cast(v8i, g1), z4, z4, z8, 0);
#else
    __builtin_amdgcn_tensor_load_to_lds(g0, __builtin_bit_cast(v8i, g1), z4, z4, 0);
#endif
  };

  auto issue_loads_a = [&](int kk) {
#pragma unroll
    for (int it = 0; it < 2; ++it) {            // A: 64 rows x 8 f4-units/row
      int u = tid + it * 256;
      int r = u >> 3, c8 = u & 7;
      const float* p = &Hs[(size_t)(rowBase + r) * Hd + kk + c8 * 8];
      ra[it][0] = *(const float4*)p;
      ra[it][1] = *(const float4*)(p + 4);
    }
  };

  issue_loads_a(0);
  if (w == 0) tdm_issue(0, 0);

  for (int kc = 0; kc < nCh; ++kc) {
    __syncthreads();   // prev chunk fully consumed: LDS a free, other B buffer free

    // DMA next B tile into the other buffer while this chunk computes
    if (w == 0 && kc + 1 < nCh) tdm_issue((kc + 1) * KB, (kc + 1) & 1);

    // pack staged A -> bf16 LDS
#pragma unroll
    for (int it = 0; it < 2; ++it) {
      int u = tid + it * 256;
      int r = u >> 3, c8 = u & 7;
      *(uint4*)&sm.g.a[r * KB2 + c8 * 4] = pk4(ra[it][0], ra[it][1]);
    }
    if (kc + 1 < nCh) issue_loads_a((kc + 1) * KB);

    // wave0 guarantees this chunk's B tile has landed, then everyone syncs
    if (w == 0) {
      if (kc + 1 < nCh) __builtin_amdgcn_s_wait_tensorcnt(1);
      else              __builtin_amdgcn_s_wait_tensorcnt(0);
    }
    __syncthreads();

    const unsigned* bb = &sm.g.b[kc & 1][0];
#pragma unroll
    for (int ks = 0; ks < KB; ks += 32) {
      v16bf af[2], bfr[2];
#pragma unroll
      for (int mi = 0; mi < 2; ++mi) {
        int row = wm * 32 + mi * 16 + (lane & 15);
        int kb  = ks + ((lane >> 4) << 4);
        const unsigned* p = &sm.g.a[row * KB2 + (kb >> 1)];
        Frag32 f; f.lo = *(const uint4*)p; f.hi = *(const uint4*)(p + 4);
        af[mi] = __builtin_bit_cast(v16bf, f);
      }
#pragma unroll
      for (int ni = 0; ni < 2; ++ni) {
        int col = wn * 32 + ni * 16 + (lane & 15);
        int kb  = ks + ((lane >> 4) << 4);
        const unsigned* p = &bb[col * KB2 + (kb >> 1)];
        Frag32 f; f.lo = *(const uint4*)p; f.hi = *(const uint4*)(p + 4);
        bfr[ni] = __builtin_bit_cast(v16bf, f);
      }
#pragma unroll
      for (int mi = 0; mi < 2; ++mi)
#pragma unroll
        for (int ni = 0; ni < 2; ++ni)
          acc[mi][ni] = __builtin_amdgcn_wmma_f32_16x16x32_bf16(
              false, af[mi], false, bfr[ni], (short)0, acc[mi][ni], false, false);
    }
  }

  __syncthreads();
#pragma unroll
  for (int ni = 0; ni < 2; ++ni) {
    int col = wn * 32 + ni * 16 + (lane & 15);
    float bv = bias[col];
#pragma unroll
    for (int mi = 0; mi < 2; ++mi) {
      int row0 = wm * 32 + mi * 16 + ((lane >> 4) << 3);
#pragma unroll
      for (int r = 0; r < 8; ++r)
        sm.logits[row0 + r][col] = acc[mi][ni][r] + bv;
    }
  }
  __syncthreads();
  topk_epilogue(sm.logits, out, rowBase, w, lane);
}

// =======================================================================
// Fallback (no-scratch) kernel: B loaded via VGPRs
// =======================================================================
union SMem {
  struct {
    alignas(16) unsigned a[BM * KB2];
    alignas(16) unsigned b[BN * KB2];
  } g;
  alignas(16) float logits[BM][BN];
};

__global__ __launch_bounds__(256) void router_topk_kernel(
    const float* __restrict__ Hs, const float* __restrict__ Wt,
    const float* __restrict__ bias, float* __restrict__ out, int Hd)
{
  __shared__ SMem sm;
  const int tid  = threadIdx.x;
  const int lane = tid & 31;
  const int w    = tid >> 5;
  const int wm   = w & 1;
  const int wn   = w >> 1;
  const int rowBase = blockIdx.x * BM;

  v8f acc[2][2] = {};
  float4 ra[2][2];
  float4 rb[4][2];
  const int nCh = Hd / KB;

  auto issue_loads = [&](int kk) {
#pragma unroll
    for (int it = 0; it < 2; ++it) {
      int u = tid + it * 256;
      int r = u >> 3, c8 = u & 7;
      const float* p = &Hs[(size_t)(rowBase + r) * Hd + kk + c8 * 8];
      ra[it][0] = *(const float4*)p;
      ra[it][1] = *(const float4*)(p + 4);
    }
#pragma unroll
    for (int it = 0; it < 4; ++it) {
      int u = tid + it * 256;
      int r = u >> 3, c8 = u & 7;
      const float* p = &Wt[(size_t)r * Hd + kk + c8 * 8];
      rb[it][0] = *(const float4*)p;
      rb[it][1] = *(const float4*)(p + 4);
    }
  };

  issue_loads(0);

  for (int kc = 0; kc < nCh; ++kc) {
    __syncthreads();
#pragma unroll
    for (int it = 0; it < 2; ++it) {
      int u = tid + it * 256;
      int r = u >> 3, c8 = u & 7;
      *(uint4*)&sm.g.a[r * KB2 + c8 * 4] = pk4(ra[it][0], ra[it][1]);
    }
#pragma unroll
    for (int it = 0; it < 4; ++it) {
      int u = tid + it * 256;
      int r = u >> 3, c8 = u & 7;
      *(uint4*)&sm.g.b[r * KB2 + c8 * 4] = pk4(rb[it][0], rb[it][1]);
    }
    __syncthreads();
    if (kc + 1 < nCh) issue_loads((kc + 1) * KB);

#pragma unroll
    for (int ks = 0; ks < KB; ks += 32) {
      v16bf af[2], bfr[2];
#pragma unroll
      for (int mi = 0; mi < 2; ++mi) {
        int row = wm * 32 + mi * 16 + (lane & 15);
        int kb  = ks + ((lane >> 4) << 4);
        const unsigned* p = &sm.g.a[row * KB2 + (kb >> 1)];
        Frag32 f; f.lo = *(const uint4*)p; f.hi = *(const uint4*)(p + 4);
        af[mi] = __builtin_bit_cast(v16bf, f);
      }
#pragma unroll
      for (int ni = 0; ni < 2; ++ni) {
        int col = wn * 32 + ni * 16 + (lane & 15);
        int kb  = ks + ((lane >> 4) << 4);
        const unsigned* p = &sm.g.b[col * KB2 + (kb >> 1)];
        Frag32 f; f.lo = *(const uint4*)p; f.hi = *(const uint4*)(p + 4);
        bfr[ni] = __builtin_bit_cast(v16bf, f);
      }
#pragma unroll
      for (int mi = 0; mi < 2; ++mi)
#pragma unroll
        for (int ni = 0; ni < 2; ++ni)
          acc[mi][ni] = __builtin_amdgcn_wmma_f32_16x16x32_bf16(
              false, af[mi], false, bfr[ni], (short)0, acc[mi][ni], false, false);
    }
  }

  __syncthreads();
#pragma unroll
  for (int ni = 0; ni < 2; ++ni) {
    int col = wn * 32 + ni * 16 + (lane & 15);
    float bv = bias[col];
#pragma unroll
    for (int mi = 0; mi < 2; ++mi) {
      int row0 = wm * 32 + mi * 16 + ((lane >> 4) << 3);
#pragma unroll
      for (int r = 0; r < 8; ++r)
        sm.logits[row0 + r][col] = acc[mi][ni][r] + bv;
    }
  }
  __syncthreads();
  topk_epilogue(sm.logits, out, rowBase, w, lane);
}

extern "C" void kernel_launch(void* const* d_in, const int* in_sizes, int n_in,
                              void* d_out, int out_size, void* d_ws, size_t ws_size,
                              hipStream_t stream) {
  (void)n_in; (void)out_size;
  const float* Hs   = (const float*)d_in[0];  // [T, Hd] fp32
  const float* Wt   = (const float*)d_in[1];  // [E, Hd] fp32
  const float* bias = (const float*)d_in[2];  // [E] fp32
  float* out = (float*)d_out;                 // [T, E] fp32

  const int E  = in_sizes[2];           // 128 (== BN)
  const int Hd = in_sizes[1] / E;       // 2880
  const int T  = in_sizes[0] / Hd;      // 8192

  const size_t wbytes = (size_t)E * (size_t)Hd * 2u; // bf16 W copy
  if (d_ws != nullptr && ws_size >= wbytes) {
    const int n8 = (E * Hd) / 8;
    cvt_w_kernel<<<(n8 + 255) / 256, 256, 0, stream>>>(Wt, (unsigned*)d_ws, n8);
    router_topk_tdm_kernel<<<dim3(T / BM), 256, 0, stream>>>(
        Hs, (const unsigned*)d_ws, bias, out, Hd);
  } else {
    router_topk_kernel<<<dim3(T / BM), 256, 0, stream>>>(Hs, Wt, bias, out, Hd);
  }
}
